// ExpertBank_87428354277650
// MI455X (gfx1250) — matrix operand order repaired and checked by reading, not compiled, not run on hardware
//
#include <hip/hip_runtime.h>
#include <math.h>

// ---- problem constants (match reference) ----
#define T_TOKENS 4096
#define DMODEL   512
#define DFF      2048
#define NEXP     8
#define KACT     2
#define TKA      (T_TOKENS * KACT)   // 8192 assignments
#define ROWS     32                  // assignments per block tile

typedef _Float16 v16h __attribute__((ext_vector_type(16)));
typedef _Float16 v8h  __attribute__((ext_vector_type(8)));
typedef _Float16 v4h  __attribute__((ext_vector_type(4)));
typedef float    v8f  __attribute__((ext_vector_type(8)));

// -------------------------------------------------------------------------
// Branch-free GELU (exact erf via Abramowitz-Stegun 7.1.26, |err|<1.5e-7).
// -------------------------------------------------------------------------
__device__ inline float gelu_f(float x) {
  float ax = fabsf(x) * 0.70710678118654752f;          // |x|/sqrt(2)
  float t  = __builtin_amdgcn_rcpf(fmaf(0.3275911f, ax, 1.0f));
  float p  = fmaf(t, 1.061405429f, -1.453152027f);
  p = fmaf(t, p, 1.421413741f);
  p = fmaf(t, p, -0.284496736f);
  p = fmaf(t, p, 0.254829592f);
  p = p * t;
  float e  = __builtin_amdgcn_exp2f(ax * ax * -1.4426950408889634f);
  float er = copysignf(fmaf(-p, e, 1.0f), x);          // erf(x/sqrt(2))
  return 0.5f * x * (1.0f + er);
}

__device__ inline void wait_async0() {
#if __has_builtin(__builtin_amdgcn_s_wait_asynccnt)
  __builtin_amdgcn_s_wait_asynccnt(0);
#else
  asm volatile("s_wait_asynccnt 0" ::: "memory");
#endif
}

// -------------------------------------------------------------------------
// Kernel 1: bucket (token,slot) assignments by expert; emit expert_loads.
// -------------------------------------------------------------------------
__global__ void moe_gather(const int* __restrict__ sel,
                           int* __restrict__ counts,
                           int* __restrict__ lists,
                           float* __restrict__ loads) {
  const int tid = threadIdx.x;
  if (tid < NEXP) counts[tid] = 0;
  __syncthreads();
  for (int i = tid; i < TKA; i += blockDim.x) {
    int e = sel[i] & 7;
    int p = atomicAdd(&counts[e], 1);
    lists[e * TKA + p] = i;                // i = token*KACT + slot
  }
  __syncthreads();
  __threadfence();
  if (tid < NEXP) loads[tid] = (float)counts[tid] * (1.0f / (float)T_TOKENS);
}

// -------------------------------------------------------------------------
// Kernel 1b: pack W^T into fragment-major f16 A-operand layout.
// A (16x32 f16) per CDNA5 7.12.2: lane L holds M = L%16 (half = L/16);
// element i -> K = half*16 + i (contiguous, as B).  One fragment = 32 lanes x 32 B
// = 1 KB contiguous -> wave loads it with two coalesced b128s.
// -------------------------------------------------------------------------
__global__ void pack_wt(const float* __restrict__ src, _Float16* __restrict__ dst,
                        int Kdim, int Mdim) {
  const int KB = Kdim >> 5, MT = Mdim >> 4;
  int gid = blockIdx.x * blockDim.x + threadIdx.x;   // (frag, lane)
  int L  = gid & 31;
  int f  = gid >> 5;
  int kb = f % KB;  int r = f / KB;
  int mt = r % MT;  int e = r / MT;
  int half = L >> 4;
  int M = mt * 16 + (L & 15);
  const float* s = src + (size_t)e * Kdim * Mdim;
  _Float16* d = dst + (size_t)gid * 16;
#pragma unroll
  for (int i = 0; i < 16; ++i) {
    int K = kb * 32 + half * 16 + i;
    d[i] = (_Float16)s[(size_t)K * Mdim + M];
  }
}

// -------------------------------------------------------------------------
// Kernel 1c: convert hidden_states to f16 (async-copy source); zero pad row.
// -------------------------------------------------------------------------
__global__ void conv_x(const float* __restrict__ X, _Float16* __restrict__ Xh,
                       _Float16* __restrict__ zbuf) {
  int gid = blockIdx.x * blockDim.x + threadIdx.x;   // one float4 each
  const float4 v = *(const float4*)(X + (size_t)gid * 4);
  v4h h = { (_Float16)v.x, (_Float16)v.y, (_Float16)v.z, (_Float16)v.w };
  *(v4h*)(Xh + (size_t)gid * 4) = h;
  if (gid < DMODEL) zbuf[gid] = (_Float16)0.0f;
}

// ---- fragment loader: 32 contiguous bytes (16B aligned) ----
__device__ inline v16h frag_from(const _Float16* p) {
  const v8h* q = (const v8h*)p;
  v8h lo = q[0], hi = q[1];
  v16h r;
#pragma unroll
  for (int i = 0; i < 8; ++i) { r[i] = lo[i]; r[i + 8] = hi[i]; }
  return r;
}
// B (32x16 f16): lane L holds N = L%16; element i -> K = (L/16)*16 + i.
// Activations stored row-major [N][K] in LDS -> 32 contiguous bytes.

// -------------------------------------------------------------------------
// Kernel 2 (fast path): fused per-expert MLP. Packed f16 weights as A
// (two global_load_b128/frag), token rows as B from LDS. X tile staged via
// async global->LDS DMA (ASYNCcnt path). 256 thr = 8 waves; 32 rows/tile.
// -------------------------------------------------------------------------
__launch_bounds__(256, 1)
__global__ void moe_ffn_packed(const _Float16* __restrict__ Xh,
                               const _Float16* __restrict__ zbuf,
                               const _Float16* __restrict__ W1t,
                               const _Float16* __restrict__ W2t,
                               const int* __restrict__ counts,
                               const int* __restrict__ lists,
                               float* __restrict__ Y) {
  __shared__ _Float16 Xs[ROWS * DMODEL];  // 32 KB
  __shared__ _Float16 Hs[ROWS * DFF];     // 128 KB

  const int e    = blockIdx.y;
  const int tile = blockIdx.x;
  const int cnt  = counts[e];
  if (tile * ROWS >= cnt) return;

  const int tid  = threadIdx.x;
  const int wave = tid >> 5;
  const int lane = tid & 31;
  const int half = lane >> 4;
  const int l16  = lane & 15;
  const int* mylist = lists + e * TKA + tile * ROWS;

  // ---- stage X tile into LDS with async DMA (16B per lane-transfer) ----
  for (int j = tid; j < ROWS * (DMODEL / 8); j += 256) {  // 64 chunks per row
    int r = j >> 6, seg = j & 63;
    const _Float16* src = (tile * ROWS + r < cnt)
        ? Xh + (size_t)(mylist[r] >> 1) * DMODEL + seg * 8
        : zbuf + seg * 8;
    unsigned lds = (unsigned)(size_t)&Xs[(size_t)r * DMODEL + seg * 8];
    unsigned long long ga = (unsigned long long)(size_t)src;
    asm volatile("global_load_async_to_lds_b128 %0, %1, off"
                 :: "v"(lds), "v"(ga) : "memory");
  }
  wait_async0();
  __syncthreads();

  // ---- stage 1: Hs = gelu(Xs @ W1[e]); A = W1^T frags, B = Xs rows ----
  {
    const _Float16* W1e = W1t + (size_t)e * DMODEL * DFF;   // 128*16 frags
    for (int t = 0; t < 16; ++t) {
      const int mt = wave * 16 + t;                          // ff col tile
      const _Float16* abase = W1e + ((size_t)mt * (DMODEL / 32)) * 512 + lane * 16;
      v8f c0 = {}; v8f c1 = {};
      for (int kb = 0; kb < DMODEL / 32; ++kb) {
        v16h a  = frag_from(abase + (size_t)kb * 512);
        v16h b0 = frag_from(&Xs[(size_t)l16        * DMODEL + kb * 32 + half * 16]);
        v16h b1 = frag_from(&Xs[(size_t)(16 + l16) * DMODEL + kb * 32 + half * 16]);
        c0 = __builtin_amdgcn_wmma_f32_16x16x32_f16(false, a, false, b0,
                                                    (short)0, c0, false, false);
        c1 = __builtin_amdgcn_wmma_f32_16x16x32_f16(false, a, false, b1,
                                                    (short)0, c1, false, false);
      }
      // D: lane N = row-in-tile = l16, element v -> M(ffcol) = mt*16 + 8*half + v
      v8h h0, h1;
#pragma unroll
      for (int v = 0; v < 8; ++v) {
        h0[v] = (_Float16)gelu_f(c0[v]);
        h1[v] = (_Float16)gelu_f(c1[v]);
      }
      *(v8h*)&Hs[(size_t)l16        * DFF + mt * 16 + 8 * half] = h0;
      *(v8h*)&Hs[(size_t)(16 + l16) * DFF + mt * 16 + 8 * half] = h1;
    }
  }
  __syncthreads();

  // ---- stage 2: Y = Hs @ W2[e]; A = W2^T frags, B = Hs rows ----
  {
    const _Float16* W2e = W2t + (size_t)e * DFF * DMODEL;   // 32*64 frags
    int a0 = (tile * ROWS + l16      < cnt) ? mylist[l16]      : -1;
    int a1 = (tile * ROWS + 16 + l16 < cnt) ? mylist[16 + l16] : -1;
    for (int t = 0; t < 4; ++t) {
      const int mt = wave * 4 + t;                           // d_model col tile
      const _Float16* abase = W2e + ((size_t)mt * (DFF / 32)) * 512 + lane * 16;
      v8f c0 = {}; v8f c1 = {};
      for (int kb = 0; kb < DFF / 32; ++kb) {
        v16h a  = frag_from(abase + (size_t)kb * 512);
        v16h b0 = frag_from(&Hs[(size_t)l16        * DFF + kb * 32 + half * 16]);
        v16h b1 = frag_from(&Hs[(size_t)(16 + l16) * DFF + kb * 32 + half * 16]);
        c0 = __builtin_amdgcn_wmma_f32_16x16x32_f16(false, a, false, b0,
                                                    (short)0, c0, false, false);
        c1 = __builtin_amdgcn_wmma_f32_16x16x32_f16(false, a, false, b1,
                                                    (short)0, c1, false, false);
      }
      const int dcol = mt * 16 + 8 * half;                   // 32B-aligned
      if (a0 >= 0) {
        float* o = Y + (size_t)a0 * DMODEL + dcol;
        *(float4*)(o)     = make_float4(c0[0], c0[1], c0[2], c0[3]);
        *(float4*)(o + 4) = make_float4(c0[4], c0[5], c0[6], c0[7]);
      }
      if (a1 >= 0) {
        float* o = Y + (size_t)a1 * DMODEL + dcol;
        *(float4*)(o)     = make_float4(c1[0], c1[1], c1[2], c1[3]);
        *(float4*)(o + 4) = make_float4(c1[4], c1[5], c1[6], c1[7]);
      }
    }
  }
}

// -------------------------------------------------------------------------
// Fallback path (small workspace): on-the-fly fp32->f16 B loads from global.
// -------------------------------------------------------------------------
__device__ inline v16h load_a_lds(const _Float16* base, int row, int ld, int k0,
                                  int half) {
  const v8h* p0 = (const v8h*)(base + (size_t)row * ld + k0 + half * 16);
  const v8h* p1 = (const v8h*)(base + (size_t)row * ld + k0 + half * 16 + 8);
  v8h lo = *p0, hi = *p1;
  v16h a;
#pragma unroll
  for (int i = 0; i < 8; ++i) { a[i] = lo[i]; a[i + 8] = hi[i]; }
  return a;
}
__device__ inline v16h load_b_global_f32(const float* __restrict__ w, int k0,
                                         int n, int ld, int half) {
  const float* bp = w + (size_t)(k0 + half * 16) * ld + n;
  v16h b;
#pragma unroll
  for (int i = 0; i < 16; ++i) b[i] = (_Float16)bp[(size_t)i * ld];
  return b;
}

__launch_bounds__(256, 1)
__global__ void moe_ffn_fallback(const float* __restrict__ X,
                                 const float* __restrict__ W1,
                                 const float* __restrict__ W2,
                                 const int* __restrict__ counts,
                                 const int* __restrict__ lists,
                                 float* __restrict__ Y) {
  __shared__ _Float16 Xs[ROWS * DMODEL];
  __shared__ _Float16 Hs[ROWS * DFF];
  const int e = blockIdx.y, tile = blockIdx.x;
  const int cnt = counts[e];
  if (tile * ROWS >= cnt) return;
  const int tid = threadIdx.x, wave = tid >> 5, lane = tid & 31;
  const int half = lane >> 4, l16 = lane & 15;
  const int* mylist = lists + e * TKA + tile * ROWS;

  for (int l = tid; l < ROWS * DMODEL; l += 256) {
    int r = l >> 9, c = l & (DMODEL - 1);
    float v = 0.0f;
    if (tile * ROWS + r < cnt) v = X[(size_t)(mylist[r] >> 1) * DMODEL + c];
    Xs[l] = (_Float16)v;
  }
  __syncthreads();
  {
    const float* W1e = W1 + (size_t)e * DMODEL * DFF;
    const int strip = wave * 256;
    for (int t = 0; t < 16; ++t) {
      const int n = strip + t * 16 + l16;
      v8f c0 = {}; v8f c1 = {};
      for (int k0 = 0; k0 < DMODEL; k0 += 32) {
        v16h b  = load_b_global_f32(W1e, k0, n, DFF, half);
        v16h a0 = load_a_lds(Xs, l16, DMODEL, k0, half);
        v16h a1 = load_a_lds(Xs, 16 + l16, DMODEL, k0, half);
        c0 = __builtin_amdgcn_wmma_f32_16x16x32_f16(false, a0, false, b, (short)0, c0, false, false);
        c1 = __builtin_amdgcn_wmma_f32_16x16x32_f16(false, a1, false, b, (short)0, c1, false, false);
      }
#pragma unroll
      for (int v = 0; v < 8; ++v) {
        Hs[(size_t)(v + 8 * half) * DFF + n]      = (_Float16)gelu_f(c0[v]);
        Hs[(size_t)(16 + v + 8 * half) * DFF + n] = (_Float16)gelu_f(c1[v]);
      }
    }
  }
  __syncthreads();
  {
    const float* W2e = W2 + (size_t)e * DFF * DMODEL;
    const int strip = wave * 64;
    for (int t = 0; t < 4; ++t) {
      const int n = strip + t * 16 + l16;
      v8f c0 = {}; v8f c1 = {};
      for (int k0 = 0; k0 < DFF; k0 += 32) {
        v16h b  = load_b_global_f32(W2e, k0, n, DMODEL, half);
        v16h a0 = load_a_lds(Hs, l16, DFF, k0, half);
        v16h a1 = load_a_lds(Hs, 16 + l16, DFF, k0, half);
        c0 = __builtin_amdgcn_wmma_f32_16x16x32_f16(false, a0, false, b, (short)0, c0, false, false);
        c1 = __builtin_amdgcn_wmma_f32_16x16x32_f16(false, a1, false, b, (short)0, c1, false, false);
      }
#pragma unroll
      for (int v = 0; v < 8; ++v) {
        int r0 = v + 8 * half;
        if (tile * ROWS + r0 < cnt) Y[(size_t)mylist[r0] * DMODEL + n] = c0[v];
        int r1 = 16 + v + 8 * half;
        if (tile * ROWS + r1 < cnt) Y[(size_t)mylist[r1] * DMODEL + n] = c1[v];
      }
    }
  }
}

// -------------------------------------------------------------------------
extern "C" void kernel_launch(void* const* d_in, const int* in_sizes, int n_in,
                              void* d_out, int out_size, void* d_ws, size_t ws_size,
                              hipStream_t stream) {
  const float* X   = (const float*)d_in[0];
  const int*   sel = (const int*)d_in[1];
  const float* W1  = (const float*)d_in[3];
  const float* W2  = (const float*)d_in[4];
  float* Y     = (float*)d_out;
  float* loads = Y + (size_t)T_TOKENS * KACT * DMODEL;

  int* counts = (int*)d_ws;                 // 8 ints
  int* lists  = counts + NEXP;              // 8*8192 ints
  const size_t W_OFF    = 262400;           // bytes, 16B aligned, past lists
  const size_t W_BYTES  = (size_t)NEXP * DMODEL * DFF * sizeof(_Float16); // 16 MB
  const size_t XH_OFF   = W_OFF + 2 * W_BYTES;
  const size_t XH_BYTES = (size_t)T_TOKENS * DMODEL * sizeof(_Float16);   // 4 MB
  const size_t Z_OFF    = XH_OFF + XH_BYTES;
  const size_t NEED     = Z_OFF + DMODEL * sizeof(_Float16);

  moe_gather<<<1, 256, 0, stream>>>(sel, counts, lists, loads);

  dim3 grid(TKA / ROWS, NEXP);
  if (ws_size >= NEED) {
    _Float16* W1t = (_Float16*)((char*)d_ws + W_OFF);
    _Float16* W2t = W1t + (size_t)NEXP * DMODEL * DFF;
    _Float16* Xh  = (_Float16*)((char*)d_ws + XH_OFF);
    _Float16* zb  = (_Float16*)((char*)d_ws + Z_OFF);
    const int pack_threads = NEXP * (DFF / 16) * (DMODEL / 32) * 32;  // 524288
    pack_wt<<<pack_threads / 256, 256, 0, stream>>>(W1, W1t, DMODEL, DFF);
    pack_wt<<<pack_threads / 256, 256, 0, stream>>>(W2, W2t, DFF, DMODEL);
    conv_x<<<(T_TOKENS * DMODEL / 4) / 256, 256, 0, stream>>>(X, Xh, zb);
    moe_ffn_packed<<<grid, 256, 0, stream>>>(Xh, zb, W1t, W2t, counts, lists, Y);
  } else {
    moe_ffn_fallback<<<grid, 256, 0, stream>>>(X, W1, W2, counts, lists, Y);
  }
}
